// VoxelPointnetSAModuleFSMSG_77713138253924
// MI455X (gfx1250) — compile-verified, matrix operand order checked by reading, not run on hardware
//
#include <hip/hip_runtime.h>
#include <hip/hip_bf16.h>
#include <stdint.h>

#define NFP   8192
#define CFEAT 64
#define NPT   1024
#define NS    32
#define R2    0.64f
#define BATCH 4

typedef __attribute__((ext_vector_type(16))) __bf16 v16bf;
typedef __attribute__((ext_vector_type(8)))  float  v8f;

union FragBF { v16bf v; unsigned int d[8]; };

__device__ __forceinline__ unsigned short f2bf(float f) {
  unsigned int u = __float_as_uint(f);
  u += 0x7FFFu + ((u >> 16) & 1u);          // round-to-nearest-even
  return (unsigned short)(u >> 16);
}

// ISA 7.12.2: 16-bit A/B fragment: dword v holds K-pair; lanes 0-15 cover
// K=[0..7]+[16..23], lanes 16-31 cover K=[8..15]+[24..31].
__device__ __forceinline__ int kpair(int v, int hi) {
  return (v < 4) ? (hi * 8 + v * 2) : (16 + hi * 8 + (v - 4) * 2);
}

__device__ __forceinline__ v16bf load_frag(const unsigned short* __restrict__ base,
                                           int row, int ldk, int kb, int lane) {
  const int hi = lane >> 4;
  FragBF f;
  const unsigned short* p = base + row * ldk + kb;
#pragma unroll
  for (int v = 0; v < 8; ++v)
    f.d[v] = *(const unsigned int*)(p + kpair(v, hi));
  return f.v;
}

// ---------------- Kernel 1: furthest point sampling (serial argmax loop) ----
__global__ __launch_bounds__(1024) void fps_kernel(const float* __restrict__ xyz,
                                                   float* __restrict__ out_newxyz) {
  const int b = blockIdx.x;
  const int t = threadIdx.x;
  const int lane = t & 31, wave = t >> 5;
  __shared__ int   sIdx[NPT];
  __shared__ float sLast[3];
  __shared__ float sRedD[32];
  __shared__ int   sRedI[32];

  const float* bx = xyz + (size_t)b * NFP * 3;
  float px[8], py[8], pz[8], dist[8];
#pragma unroll
  for (int k = 0; k < 8; ++k) {
    int p = t + k * 1024;
    px[k] = bx[p * 3 + 0]; py[k] = bx[p * 3 + 1]; pz[k] = bx[p * 3 + 2];
    dist[k] = 1e10f;
  }
  if (t == 0) { sLast[0] = px[0]; sLast[1] = py[0]; sLast[2] = pz[0]; sIdx[0] = 0; }
  __syncthreads();

  for (int i = 1; i < NPT; ++i) {
    const float lx = sLast[0], ly = sLast[1], lz = sLast[2];
    float bestd = -1.0f; int bestp = 0;
#pragma unroll
    for (int k = 0; k < 8; ++k) {
      float dx = px[k] - lx, dy = py[k] - ly, dz = pz[k] - lz;
      float nd = fminf(dist[k], dx * dx + dy * dy + dz * dz);
      dist[k] = nd;
      if (nd > bestd) { bestd = nd; bestp = t + k * 1024; }   // ascending p: first max wins
    }
#pragma unroll
    for (int off = 1; off < 32; off <<= 1) {                  // wave32 butterfly argmax
      float od = __shfl_xor(bestd, off);
      int   oi = __shfl_xor(bestp, off);
      if (od > bestd || (od == bestd && oi < bestp)) { bestd = od; bestp = oi; }
    }
    if (lane == 0) { sRedD[wave] = bestd; sRedI[wave] = bestp; }
    __syncthreads();
    if (wave == 0) {
      bestd = sRedD[lane]; bestp = sRedI[lane];
#pragma unroll
      for (int off = 1; off < 32; off <<= 1) {
        float od = __shfl_xor(bestd, off);
        int   oi = __shfl_xor(bestp, off);
        if (od > bestd || (od == bestd && oi < bestp)) { bestd = od; bestp = oi; }
      }
      if (lane == 0) sIdx[i] = bestp;
      if (lane < 3) sLast[lane] = bx[bestp * 3 + lane];       // L2 hit: xyz fits easily
    }
    __syncthreads();
  }
  const int id = sIdx[t];
  float* o = out_newxyz + (size_t)b * NPT * 3 + t * 3;
  o[0] = bx[id * 3 + 0]; o[1] = bx[id * 3 + 1]; o[2] = bx[id * 3 + 2];
}

// ---------------- Kernel 2: ball query (one wave per sampled point) ---------
__global__ __launch_bounds__(256) void ballq_kernel(const float* __restrict__ xyz,
                                                    const float* __restrict__ newxyz,
                                                    int* __restrict__ idxb,
                                                    int* __restrict__ flagb) {
  const int lane = threadIdx.x & 31;
  const int gw = blockIdx.x * 8 + (threadIdx.x >> 5);
  const int b = gw / NPT, s = gw % NPT;
  const float* bx = xyz + (size_t)b * NFP * 3;
  const float cx = newxyz[(size_t)b * NPT * 3 + s * 3 + 0];
  const float cy = newxyz[(size_t)b * NPT * 3 + s * 3 + 1];
  const float cz = newxyz[(size_t)b * NPT * 3 + s * 3 + 2];
  int* out = idxb + ((size_t)b * NPT + s) * NS;

  int count = 0, firstIdx = NFP - 1;
  for (int j0 = 0; j0 < NFP && count < NS; j0 += 32) {
    const int j = j0 + lane;
    float dx = bx[j * 3 + 0] - cx, dy = bx[j * 3 + 1] - cy, dz = bx[j * 3 + 2] - cz;
    const bool pred = (dx * dx + dy * dy + dz * dz) < R2;
    const unsigned int mask = (unsigned int)__ballot(pred);
    if (count == 0 && mask) firstIdx = j0 + (__ffs((int)mask) - 1);
    const int pos = count + __popc(mask & ((1u << lane) - 1u));
    if (pred && pos < NS) out[pos] = j;                       // ascending = smallest-32
    count += __popc(mask);
  }
  count = min(count, NS);
  if (lane >= count) out[lane] = firstIdx;                    // pad with first (8191 if none)
  if (lane == 0) flagb[b * NPT + s] = (count > 0) ? 1 : 0;
}

// ---------------- Kernel 3: gather + MLP(WMMA bf16) + maxpool ---------------
#define SMEM_BYTES (256*96*2 + 256*64*2 + 64*96*2 + 128*64*2 + 8*128*4)  // 114688

__global__ __launch_bounds__(256) void mlp_kernel(const float* __restrict__ xyz,
                                                  const float* __restrict__ feat,
                                                  const float* __restrict__ w1,
                                                  const float* __restrict__ b1,
                                                  const float* __restrict__ w2,
                                                  const float* __restrict__ b2,
                                                  const float* __restrict__ newxyz,
                                                  const int* __restrict__ idxb,
                                                  const int* __restrict__ flagb,
                                                  float* __restrict__ outf) {
  extern __shared__ __align__(16) char smem[];
  unsigned short* Xs  = (unsigned short*)smem;        // 256 cols x 96 (K padded)
  unsigned short* H1s = Xs + 256 * 96;                // 256 cols x 64
  unsigned short* W1s = H1s + 256 * 64;               // 64 rows x 96
  unsigned short* W2s = W1s + 64 * 96;                // 128 rows x 64
  unsigned int*   Mx  = (unsigned int*)(W2s + 128 * 64);  // 8 s x 128 rows (f32 bits)

  const int tid = threadIdx.x;
  const int lane = tid & 31, wave = tid >> 5;
  const int b = blockIdx.x >> 7;
  const int s0 = (blockIdx.x & 127) * 8;

  // stage 0: gather grouped input (bf16), stage weights, init max accumulators
  {
    const int col = tid, s_l = col >> 5, n = col & 31;
    const int s = s0 + s_l;
    const int j = idxb[((size_t)b * NPT + s) * NS + n];
    const int fl = flagb[b * NPT + s];
    unsigned short* xc = Xs + col * 96;
    if (fl) {
      const float* c0 = newxyz + (size_t)b * NPT * 3 + s * 3;
      const float* pj = xyz + (size_t)b * NFP * 3 + j * 3;
      xc[0] = f2bf(pj[0] - c0[0]); xc[1] = f2bf(pj[1] - c0[1]); xc[2] = f2bf(pj[2] - c0[2]);
      const float* fb = feat + (size_t)b * CFEAT * NFP + j;
#pragma unroll 8
      for (int c = 0; c < CFEAT; ++c) xc[3 + c] = f2bf(fb[(size_t)c * NFP]);
    } else {
      for (int c = 0; c < 67; ++c) xc[c] = 0;
    }
    for (int c = 67; c < 96; ++c) xc[c] = 0;
  }
  for (int i = tid; i < 64 * 96; i += 256) {
    const int k = i % 96;
    W1s[i] = (k < 67) ? f2bf(w1[(i / 96) * 67 + k]) : (unsigned short)0;
  }
  for (int i = tid; i < 128 * 64; i += 256) W2s[i] = f2bf(w2[i]);
  for (int i = tid; i < 8 * 128; i += 256) Mx[i] = 0u;        // ReLU output >= 0
  __syncthreads();

  // layer 1: 64x256 = 4x16 tiles, K = 96 -> 3 WMMAs per tile
  for (int t = wave; t < 64; t += 8) {
    const int mt = t & 3, ct = t >> 2;
    v8f acc = {0.f, 0.f, 0.f, 0.f, 0.f, 0.f, 0.f, 0.f};
    const int mrow = mt * 16 + (lane & 15);
    const int col  = ct * 16 + (lane & 15);
#pragma unroll
    for (int kb = 0; kb < 96; kb += 32) {
      v16bf a  = load_frag(W1s, mrow, 96, kb, lane);
      v16bf bm = load_frag(Xs,  col,  96, kb, lane);
      acc = __builtin_amdgcn_wmma_f32_16x16x32_bf16(false, a, false, bm,
                                                    (short)0, acc, false, false);
    }
    const int hi = lane >> 4;
#pragma unroll
    for (int v = 0; v < 8; ++v) {
      const int m = mt * 16 + v + 8 * hi;
      H1s[col * 64 + m] = f2bf(fmaxf(acc[v] + b1[m], 0.f));
    }
  }
  __syncthreads();

  // layer 2: 128x256 = 8x16 tiles, K = 64 -> 2 WMMAs; fused max over nsample
  for (int t = wave; t < 128; t += 8) {
    const int mt = t & 7, ct = t >> 3;
    v8f acc = {0.f, 0.f, 0.f, 0.f, 0.f, 0.f, 0.f, 0.f};
    const int mrow = mt * 16 + (lane & 15);
    const int col  = ct * 16 + (lane & 15);
#pragma unroll
    for (int kb = 0; kb < 64; kb += 32) {
      v16bf a  = load_frag(W2s, mrow, 64, kb, lane);
      v16bf bm = load_frag(H1s, col,  64, kb, lane);
      acc = __builtin_amdgcn_wmma_f32_16x16x32_bf16(false, a, false, bm,
                                                    (short)0, acc, false, false);
    }
    const int hi = lane >> 4;
    const int s_l = ct >> 1;                                   // 16-col tile = half of one s
#pragma unroll
    for (int v = 0; v < 8; ++v) {
      const int m = mt * 16 + v + 8 * hi;
      float x = fmaxf(acc[v] + b2[m], 0.f);
#pragma unroll
      for (int off = 1; off < 16; off <<= 1) x = fmaxf(x, __shfl_xor(x, off));
      if ((lane & 15) == 0)                                    // uint max == float max for >=0
        atomicMax(&Mx[s_l * 128 + m], __float_as_uint(x));
    }
  }
  __syncthreads();

  for (int i = tid; i < 8 * 128; i += 256) {
    const int s_l = i >> 7, m = i & 127;
    outf[(size_t)b * 128 * NPT + m * NPT + (s0 + s_l)] = __uint_as_float(Mx[i]);
  }
}

// ---------------------------------------------------------------------------
extern "C" void kernel_launch(void* const* d_in, const int* in_sizes, int n_in,
                              void* d_out, int out_size, void* d_ws, size_t ws_size,
                              hipStream_t stream) {
  const float* xyz  = (const float*)d_in[0];
  const float* feat = (const float*)d_in[1];
  const float* w1   = (const float*)d_in[2];
  const float* b1   = (const float*)d_in[3];
  const float* w2   = (const float*)d_in[4];
  const float* b2   = (const float*)d_in[5];
  float* out        = (float*)d_out;
  float* out_newxyz = out;                        // (B,1024,3)
  float* out_feats  = out + BATCH * NPT * 3;      // (B,128,1024)

  int* idxb  = (int*)d_ws;                        // B*NPT*NS ints
  int* flagb = idxb + BATCH * NPT * NS;           // B*NPT ints

  fps_kernel<<<BATCH, 1024, 0, stream>>>(xyz, out_newxyz);
  ballq_kernel<<<(BATCH * NPT) / 8, 256, 0, stream>>>(xyz, out_newxyz, idxb, flagb);

  (void)hipFuncSetAttribute(reinterpret_cast<const void*>(mlp_kernel),
                            hipFuncAttributeMaxDynamicSharedMemorySize, SMEM_BYTES);
  mlp_kernel<<<BATCH * (NPT / 8), 256, SMEM_BYTES, stream>>>(
      xyz, feat, w1, b1, w2, b2, out_newxyz, idxb, flagb, out_feats);
}
